// FluxIPAdapterSingleAttnProcessor_54219667145158
// MI455X (gfx1250) — matrix operand order repaired and checked by reading, compiled, not run on hardware
//
#include <hip/hip_runtime.h>
#include <hip/hip_bf16.h>

// ---------------------------------------------------------------------------
// FLUX IP-Adapter single attention processor for MI455X (gfx1250, wave32).
// All GEMMs and both attentions run on v_wmma_f32_16x16x32_f16.
// Fast path: one-shot f32->f16 conversion, then async double-buffered
// global_load_async_to_lds_b128 staging in the GEMM hot loop.
// ---------------------------------------------------------------------------

#define HEADS 24
#define HEAD_DIM 128
#define SEQ 2048
#define HID 3072
#define CAD 4096
#define IPT 128

typedef __attribute__((ext_vector_type(16))) _Float16 v16h;
typedef __attribute__((ext_vector_type(8)))  float    v8f;
typedef __attribute__((ext_vector_type(4)))  unsigned int u32x4;

union Frag {            // 16 f16 values = 8 VGPRs (one WMMA A/B operand)
    v16h  h;
    u32x4 u[2];
};

__device__ __forceinline__ v8f wmma_f16(const Frag& a, const Frag& b, v8f c) {
    return __builtin_amdgcn_wmma_f32_16x16x32_f16(
        /*neg_a=*/false, a.h, /*neg_b=*/false, b.h,
        /*c_mod=*/(short)0, c, /*reuse_a=*/false, /*reuse_b=*/false);
}

// per-lane async DMA: 16 bytes global -> LDS, tracked by ASYNCcnt
__device__ __forceinline__ void async_cp16(unsigned lds_off, const _Float16* gsrc) {
    asm volatile("global_load_async_to_lds_b128 %0, %1, off"
                 :: "v"(lds_off), "v"(gsrc) : "memory");
}

// ===========================================================================
// Shared GEMM epilogue over a 128x128 head tile held in register D-layout.
// Modes: 0 q/k (+bias, rmsnorm(w), RoPE -> rows f16)
//        1 v   (+bias -> transposed f16)
//        2 ipk (rmsnorm -> rows f16)
//        3 ipv (rmsnorm -> transposed f16)
// ===========================================================================
__device__ __forceinline__ void head_epilogue(
    float* lC, const v8f* acc, const float* bias, int M, int mode, float eps,
    const float* normw, const float* cosp, const float* sinp,
    _Float16* outRows, _Float16* outT, int outTstride, int tileM, int head)
{
    const int t     = threadIdx.x;
    const int wave  = t >> 5;
    const int lane  = t & 31;
    const int nlane = lane & 15;
    const int h     = lane >> 4;
    const int tileN = head * HEAD_DIM;

    float bval[8];
    #pragma unroll
    for (int dt = 0; dt < 8; ++dt)
        bval[dt] = bias ? bias[tileN + 16 * dt + nlane] : 0.f;

    for (int p = 0; p < 2; ++p) {          // two 64-row passes through lC
        __syncthreads();
        if ((wave >> 2) == p) {
            const int lw = wave & 3;
            #pragma unroll
            for (int dt = 0; dt < 8; ++dt)
                #pragma unroll
                for (int r = 0; r < 8; ++r)
                    lC[(16 * lw + 8 * h + r) * 132 + 16 * dt + nlane] = acc[dt][r] + bval[dt];
        }
        __syncthreads();

        if (mode == 0 || mode == 2) {
            if (t < 64) {
                const int sIdx = tileM + p * 64 + t;
                const float* cr = &lC[t * 132];
                float ss = 0.f;
                for (int d = 0; d < HEAD_DIM; ++d) { float x = cr[d]; ss += x * x; }
                const float rs = rsqrtf(ss * (1.f / HEAD_DIM) + eps);
                _Float16* orow = outRows + ((size_t)head * M + sIdx) * HEAD_DIM;
                if (mode == 0) {
                    const float* cRow = cosp + (size_t)sIdx * HEAD_DIM;
                    const float* sRow = sinp + (size_t)sIdx * HEAD_DIM;
                    for (int i = 0; i < HEAD_DIM / 2; ++i) {
                        const int d0 = 2 * i, d1 = 2 * i + 1;
                        const float x0 = cr[d0] * rs * normw[d0];
                        const float x1 = cr[d1] * rs * normw[d1];
                        orow[d0] = (_Float16)(x0 * cRow[d0] - x1 * sRow[d0]);
                        orow[d1] = (_Float16)(x1 * cRow[d1] + x0 * sRow[d1]);
                    }
                } else {
                    for (int d = 0; d < HEAD_DIM; ++d)
                        orow[d] = (_Float16)(cr[d] * rs);
                }
            }
        } else {
            if (mode == 3 && t < 64) {     // normalize rows in place first
                float* cr = &lC[t * 132];
                float ss = 0.f;
                for (int d = 0; d < HEAD_DIM; ++d) { float x = cr[d]; ss += x * x; }
                const float rs = rsqrtf(ss * (1.f / HEAD_DIM) + eps);
                for (int d = 0; d < HEAD_DIM; ++d) cr[d] *= rs;
            }
            if (mode == 3) __syncthreads();
            // transposed f16 store: outT[head][d][token]
            const int d  = t & 127;
            const int th = t >> 7;
            _Float16* od = outT + ((size_t)head * HEAD_DIM + d) * outTstride
                                + tileM + p * 64 + th * 32;
            for (int j = 0; j < 32; ++j)
                od[j] = (_Float16)lC[(th * 32 + j) * 132 + d];
        }
    }
}

// ===========================================================================
// one-shot f32 -> f16 conversion (8 elements / thread, 16B stores)
// ===========================================================================
__global__ __launch_bounds__(256) void cvt_f32_f16_kernel(
    const float* __restrict__ src, _Float16* __restrict__ dst, int n)
{
    const int i = (blockIdx.x * 256 + threadIdx.x) * 8;
    if (i >= n) return;
    const float4 a = *(const float4*)(src + i);
    const float4 b = *(const float4*)(src + i + 4);
    union { _Float16 hh[8]; u32x4 u; } o;
    o.hh[0]=(_Float16)a.x; o.hh[1]=(_Float16)a.y; o.hh[2]=(_Float16)a.z; o.hh[3]=(_Float16)a.w;
    o.hh[4]=(_Float16)b.x; o.hh[5]=(_Float16)b.y; o.hh[6]=(_Float16)b.z; o.hh[7]=(_Float16)b.w;
    *(u32x4*)(dst + i) = o.u;
}

// ===========================================================================
// FAST PATH GEMM: f16 inputs, double-buffered async LDS staging.
// C[tileM..+128][head*128..+128] = X16[M x K] * W16^T  (W16 is [3072][K])
// ===========================================================================
__global__ __launch_bounds__(256) void gemm_head_async_kernel(
    const _Float16* __restrict__ X16, const _Float16* __restrict__ W16,
    const float* __restrict__ bias, int M, int K, int mode, float eps,
    const float* __restrict__ normw,
    const float* __restrict__ cosp, const float* __restrict__ sinp,
    _Float16* __restrict__ outRows, _Float16* __restrict__ outT, int outTstride)
{
    // LDS: 2 x (A 8KB + B 8KB) double buffer, aliased with epilogue lC
    __shared__ __align__(16) char smem[64 * 132 * 4];   // 33792 B >= 32768 B
    float* lC = (float*)smem;

    const int t     = threadIdx.x;
    const int wave  = t >> 5;
    const int lane  = t & 31;
    const int nlane = lane & 15;
    const int h     = lane >> 4;

    const int tileM = blockIdx.x * 128;
    const int head  = blockIdx.y;
    const int tileN = head * HEAD_DIM;

    v8f acc[8];
    #pragma unroll
    for (int i = 0; i < 8; ++i)
        #pragma unroll
        for (int r = 0; r < 8; ++r) acc[i][r] = 0.f;

    const int srow = t >> 1;            // staged row 0..127
    const int kb2  = (t & 1);           // 0/1 -> 32B halves of the 64B row
    const unsigned ldsBase = (unsigned)(size_t)smem;

    // issue one 8KB A tile + 8KB B tile into buffer b (4 async b128 / thread)
    auto issue = [&](int b, int kk) {
        const unsigned la = ldsBase + (unsigned)b * 16384u + srow * 64u + kb2 * 32u;
        const unsigned lb = la + 8192u;
        const _Float16* ga = X16 + (size_t)(tileM + srow) * K + kk + kb2 * 16;
        const _Float16* gb = W16 + (size_t)(tileN + srow) * K + kk + kb2 * 16;
        async_cp16(la,      ga);
        async_cp16(la + 16, ga + 8);
        async_cp16(lb,      gb);
        async_cp16(lb + 16, gb + 8);
    };

    issue(0, 0);
    int cur = 0;
    for (int kk = 0; kk < K; kk += 32) {
        if (kk + 32 < K) {
            issue(cur ^ 1, kk + 32);                       // overlap next DMA
            asm volatile("s_wait_asynccnt 0x4" ::: "memory");  // current tile landed
        } else {
            asm volatile("s_wait_asynccnt 0x0" ::: "memory");
        }
        __syncthreads();                                    // all waves' slices visible

        const _Float16* lA = (const _Float16*)(smem + cur * 16384);
        const _Float16* lB = lA + 4096;
        const int mrow = 16 * wave + nlane;
        Frag a;
        a.u[0] = *(const u32x4*)(lA + mrow * 32 + 8 * h);
        a.u[1] = *(const u32x4*)(lA + mrow * 32 + 16 + 8 * h);
        #pragma unroll
        for (int nt = 0; nt < 8; ++nt) {
            const int n = 16 * nt + nlane;
            Frag b;
            b.u[0] = *(const u32x4*)(lB + n * 32 + 8 * h);
            b.u[1] = *(const u32x4*)(lB + n * 32 + 16 + 8 * h);
            acc[nt] = wmma_f16(a, b, acc[nt]);
        }
        __syncthreads();        // everyone done reading `cur` before it is refilled
        cur ^= 1;
    }

    head_epilogue(lC, acc, bias, M, mode, eps, normw, cosp, sinp,
                  outRows, outT, outTstride, tileM, head);
}

// ===========================================================================
// FALLBACK GEMM (small workspace): f32 inputs, inline f32->f16 staging.
// ===========================================================================
__global__ __launch_bounds__(256) void gemm_head_kernel(
    const float* __restrict__ X, const float* __restrict__ W,
    const float* __restrict__ bias, int M, int K, int mode, float eps,
    const float* __restrict__ normw,
    const float* __restrict__ cosp, const float* __restrict__ sinp,
    _Float16* __restrict__ outRows, _Float16* __restrict__ outT, int outTstride)
{
    __shared__ __align__(16) char smem[64 * 132 * 4];
    _Float16* lA = (_Float16*)smem;
    _Float16* lB = (_Float16*)(smem + 8192);
    float*    lC = (float*)smem;

    const int t     = threadIdx.x;
    const int wave  = t >> 5;
    const int lane  = t & 31;
    const int nlane = lane & 15;
    const int h     = lane >> 4;

    const int tileM = blockIdx.x * 128;
    const int head  = blockIdx.y;
    const int tileN = head * HEAD_DIM;

    v8f acc[8];
    #pragma unroll
    for (int i = 0; i < 8; ++i)
        #pragma unroll
        for (int r = 0; r < 8; ++r) acc[i][r] = 0.f;

    const int srow = t >> 1;
    const int ko   = (t & 1) * 16;

    for (int kk = 0; kk < K; kk += 32) {
        __syncthreads();
        {
            const float4* sa = (const float4*)(X + (size_t)(tileM + srow) * K + kk + ko);
            const float4* sb = (const float4*)(W + (size_t)(tileN + srow) * K + kk + ko);
            if (kk + 32 < K) {
                __builtin_prefetch(X + (size_t)(tileM + srow) * K + kk + 32 + ko, 0, 1);
                __builtin_prefetch(W + (size_t)(tileN + srow) * K + kk + 32 + ko, 0, 1);
            }
            float4 a0 = sa[0], a1 = sa[1], a2 = sa[2], a3 = sa[3];
            float4 b0 = sb[0], b1 = sb[1], b2 = sb[2], b3 = sb[3];
            _Float16* da = lA + srow * 32 + ko;
            _Float16* db = lB + srow * 32 + ko;
            da[0]=(_Float16)a0.x; da[1]=(_Float16)a0.y; da[2]=(_Float16)a0.z; da[3]=(_Float16)a0.w;
            da[4]=(_Float16)a1.x; da[5]=(_Float16)a1.y; da[6]=(_Float16)a1.z; da[7]=(_Float16)a1.w;
            da[8]=(_Float16)a2.x; da[9]=(_Float16)a2.y; da[10]=(_Float16)a2.z; da[11]=(_Float16)a2.w;
            da[12]=(_Float16)a3.x; da[13]=(_Float16)a3.y; da[14]=(_Float16)a3.z; da[15]=(_Float16)a3.w;
            db[0]=(_Float16)b0.x; db[1]=(_Float16)b0.y; db[2]=(_Float16)b0.z; db[3]=(_Float16)b0.w;
            db[4]=(_Float16)b1.x; db[5]=(_Float16)b1.y; db[6]=(_Float16)b1.z; db[7]=(_Float16)b1.w;
            db[8]=(_Float16)b2.x; db[9]=(_Float16)b2.y; db[10]=(_Float16)b2.z; db[11]=(_Float16)b2.w;
            db[12]=(_Float16)b3.x; db[13]=(_Float16)b3.y; db[14]=(_Float16)b3.z; db[15]=(_Float16)b3.w;
        }
        __syncthreads();
        const int mrow = 16 * wave + nlane;
        Frag a;
        a.u[0] = *(const u32x4*)(lA + mrow * 32 + 8 * h);
        a.u[1] = *(const u32x4*)(lA + mrow * 32 + 16 + 8 * h);
        #pragma unroll
        for (int nt = 0; nt < 8; ++nt) {
            const int n = 16 * nt + nlane;
            Frag b;
            b.u[0] = *(const u32x4*)(lB + n * 32 + 8 * h);
            b.u[1] = *(const u32x4*)(lB + n * 32 + 16 + 8 * h);
            acc[nt] = wmma_f16(a, b, acc[nt]);
        }
    }

    head_epilogue(lC, acc, bias, M, mode, eps, normw, cosp, sinp,
                  outRows, outT, outTstride, tileM, head);
}

// ===========================================================================
// Flash attention: one wave = 16 queries x full head_dim; kv blocks of 32.
// qf:[H][S][128]  kf:[H][L][128]  vT:[H][128][L]  out:[S][3072] (f32)
// ===========================================================================
__global__ __launch_bounds__(256) void attn_kernel(
    const _Float16* __restrict__ qf, const _Float16* __restrict__ kf,
    const _Float16* __restrict__ vT, float* __restrict__ out,
    int S, int L, int accumulate)
{
    __shared__ _Float16 pbuf[8][16 * 32];   // per-wave P transpose buffer

    const int t     = threadIdx.x;
    const int wave  = t >> 5;
    const int lane  = t & 31;
    const int nlane = lane & 15;
    const int h     = lane >> 4;

    const int head = blockIdx.y;
    const int qb   = blockIdx.x * 128 + wave * 16;

    const float kscale = 0.08838834764831845f * 1.44269504088896340736f; // 1/sqrt(128)*log2(e)

    Frag qa[4];
    {
        const _Float16* qrow = qf + ((size_t)head * S + (qb + nlane)) * HEAD_DIM;
        #pragma unroll
        for (int c = 0; c < 4; ++c) {
            qa[c].u[0] = *(const u32x4*)(qrow + 32 * c + 8 * h);
            qa[c].u[1] = *(const u32x4*)(qrow + 32 * c + 16 + 8 * h);
        }
    }

    v8f acc[8];
    float mrow[8], lrow[8];
    #pragma unroll
    for (int i = 0; i < 8; ++i) {
        mrow[i] = -1e30f; lrow[i] = 0.f;
        #pragma unroll
        for (int r = 0; r < 8; ++r) acc[i][r] = 0.f;
    }

    const _Float16* kbase = kf + (size_t)head * L * HEAD_DIM;
    const _Float16* vbase = vT + (size_t)head * HEAD_DIM * L;
    _Float16* pb = pbuf[wave];

    for (int kb = 0; kb < L; kb += 32) {
        v8f s0, s1;
        #pragma unroll
        for (int r = 0; r < 8; ++r) { s0[r] = 0.f; s1[r] = 0.f; }
        #pragma unroll
        for (int c = 0; c < 4; ++c) {
            const _Float16* k0 = kbase + (size_t)(kb + nlane) * HEAD_DIM + 32 * c + 8 * h;
            const _Float16* k1 = k0 + 16 * HEAD_DIM;
            Frag b0, b1;
            b0.u[0] = *(const u32x4*)(k0);     b0.u[1] = *(const u32x4*)(k0 + 16);
            b1.u[0] = *(const u32x4*)(k1);     b1.u[1] = *(const u32x4*)(k1 + 16);
            s0 = wmma_f16(qa[c], b0, s0);
            s1 = wmma_f16(qa[c], b1, s1);
        }

        float p0[8], p1[8];
        #pragma unroll
        for (int r = 0; r < 8; ++r) {
            s0[r] *= kscale; s1[r] *= kscale;
            float mx = fmaxf(s0[r], s1[r]);
            mx = fmaxf(mx, __shfl_xor(mx, 1, 32));
            mx = fmaxf(mx, __shfl_xor(mx, 2, 32));
            mx = fmaxf(mx, __shfl_xor(mx, 4, 32));
            mx = fmaxf(mx, __shfl_xor(mx, 8, 32));
            const float mnew  = fmaxf(mrow[r], mx);
            const float alpha = exp2f(mrow[r] - mnew);
            p0[r] = exp2f(s0[r] - mnew);
            p1[r] = exp2f(s1[r] - mnew);
            float rsum = p0[r] + p1[r];
            rsum += __shfl_xor(rsum, 1, 32);
            rsum += __shfl_xor(rsum, 2, 32);
            rsum += __shfl_xor(rsum, 4, 32);
            rsum += __shfl_xor(rsum, 8, 32);
            lrow[r] = lrow[r] * alpha + rsum;
            mrow[r] = mnew;
            #pragma unroll
            for (int dt = 0; dt < 8; ++dt) acc[dt][r] *= alpha;
        }

        #pragma unroll
        for (int r = 0; r < 8; ++r) {
            pb[(8 * h + r) * 32 + nlane]      = (_Float16)p0[r];
            pb[(8 * h + r) * 32 + 16 + nlane] = (_Float16)p1[r];
        }
        asm volatile("s_wait_dscnt 0" ::: "memory");
        Frag pa;
        pa.u[0] = *(const u32x4*)(pb + nlane * 32 + 8 * h);
        pa.u[1] = *(const u32x4*)(pb + nlane * 32 + 16 + 8 * h);

        #pragma unroll
        for (int dt = 0; dt < 8; ++dt) {
            const _Float16* vp = vbase + (size_t)(16 * dt + nlane) * L + kb + 8 * h;
            Frag b;
            b.u[0] = *(const u32x4*)(vp);
            b.u[1] = *(const u32x4*)(vp + 16);
            acc[dt] = wmma_f16(pa, b, acc[dt]);
        }
    }

    #pragma unroll
    for (int r = 0; r < 8; ++r) {
        const float inv = 1.f / lrow[r];
        const int sIdx  = qb + 8 * h + r;
        #pragma unroll
        for (int dt = 0; dt < 8; ++dt) {
            const size_t idx = (size_t)sIdx * HID + head * HEAD_DIM + 16 * dt + nlane;
            const float v = acc[dt][r] * inv;
            if (accumulate) out[idx] += v;
            else            out[idx]  = v;
        }
    }
}

// ===========================================================================
extern "C" void kernel_launch(void* const* d_in, const int* in_sizes, int n_in,
                              void* d_out, int out_size, void* d_ws, size_t ws_size,
                              hipStream_t stream) {
    const float* hidden  = (const float*)d_in[0];
    const float* cosp    = (const float*)d_in[1];
    const float* sinp    = (const float*)d_in[2];
    const float* ip_hid  = (const float*)d_in[3];
    const float* Wq      = (const float*)d_in[4];
    const float* bq      = (const float*)d_in[5];
    const float* Wk      = (const float*)d_in[6];
    const float* bk      = (const float*)d_in[7];
    const float* Wv      = (const float*)d_in[8];
    const float* bv      = (const float*)d_in[9];
    const float* nqw     = (const float*)d_in[10];
    const float* nkw     = (const float*)d_in[11];
    const float* Wk_ip   = (const float*)d_in[12];
    const float* Wv_ip   = (const float*)d_in[13];
    float* out = (float*)d_out;

    char* ws = (char*)d_ws;
    const size_t qkvB = (size_t)HEADS * SEQ * HEAD_DIM * sizeof(_Float16); // 12.58 MB
    const size_t ipB  = (size_t)HEADS * IPT * HEAD_DIM * sizeof(_Float16);

    size_t off = 0;
    _Float16* qf   = (_Float16*)(ws + off); off += qkvB;
    _Float16* kf   = (_Float16*)(ws + off); off += qkvB;
    _Float16* vT   = (_Float16*)(ws + off); off += qkvB;
    _Float16* ipk  = (_Float16*)(ws + off); off += ipB;
    _Float16* ipvT = (_Float16*)(ws + off); off += ipB;

    // f16 mirrors for the async fast path
    const int nHid  = SEQ * HID;         // 6,291,456
    const int nW    = HID * HID;         // 9,437,184
    const int nIph  = IPT * CAD;         // 524,288
    const int nWip  = HID * CAD;         // 12,582,912
    _Float16* hid16  = (_Float16*)(ws + off); off += (size_t)nHid * 2;
    _Float16* wq16   = (_Float16*)(ws + off); off += (size_t)nW * 2;
    _Float16* wk16   = (_Float16*)(ws + off); off += (size_t)nW * 2;
    _Float16* wv16   = (_Float16*)(ws + off); off += (size_t)nW * 2;
    _Float16* iph16  = (_Float16*)(ws + off); off += (size_t)nIph * 2;
    _Float16* wkip16 = (_Float16*)(ws + off); off += (size_t)nWip * 2;
    _Float16* wvip16 = (_Float16*)(ws + off); off += (size_t)nWip * 2;
    const bool fast = (ws_size >= off);

    dim3 blk(256);
    dim3 gQKV(SEQ / 128, HEADS);
    dim3 gIP(IPT / 128, HEADS);
    dim3 gAttn(SEQ / 128, HEADS);

    if (fast) {
        auto cvt = [&](const float* s, _Float16* d, int n) {
            cvt_f32_f16_kernel<<<dim3((n / 8 + 255) / 256), blk, 0, stream>>>(s, d, n);
        };
        cvt(hidden, hid16, nHid);
        cvt(Wq, wq16, nW);  cvt(Wk, wk16, nW);  cvt(Wv, wv16, nW);
        cvt(ip_hid, iph16, nIph);
        cvt(Wk_ip, wkip16, nWip);  cvt(Wv_ip, wvip16, nWip);

        gemm_head_async_kernel<<<gQKV, blk, 0, stream>>>(hid16, wq16, bq, SEQ, HID, 0, 1e-6f,
                                                         nqw, cosp, sinp, qf, nullptr, 0);
        gemm_head_async_kernel<<<gQKV, blk, 0, stream>>>(hid16, wk16, bk, SEQ, HID, 0, 1e-6f,
                                                         nkw, cosp, sinp, kf, nullptr, 0);
        gemm_head_async_kernel<<<gQKV, blk, 0, stream>>>(hid16, wv16, bv, SEQ, HID, 1, 0.f,
                                                         nullptr, nullptr, nullptr, nullptr, vT, SEQ);
        gemm_head_async_kernel<<<gIP, blk, 0, stream>>>(iph16, wkip16, nullptr, IPT, CAD, 2, 1e-5f,
                                                        nullptr, nullptr, nullptr, ipk, nullptr, 0);
        gemm_head_async_kernel<<<gIP, blk, 0, stream>>>(iph16, wvip16, nullptr, IPT, CAD, 3, 1e-5f,
                                                        nullptr, nullptr, nullptr, nullptr, ipvT, IPT);
    } else {
        gemm_head_kernel<<<gQKV, blk, 0, stream>>>(hidden, Wq, bq, SEQ, HID, 0, 1e-6f,
                                                   nqw, cosp, sinp, qf, nullptr, 0);
        gemm_head_kernel<<<gQKV, blk, 0, stream>>>(hidden, Wk, bk, SEQ, HID, 0, 1e-6f,
                                                   nkw, cosp, sinp, kf, nullptr, 0);
        gemm_head_kernel<<<gQKV, blk, 0, stream>>>(hidden, Wv, bv, SEQ, HID, 1, 0.f,
                                                   nullptr, nullptr, nullptr, nullptr, vT, SEQ);
        gemm_head_kernel<<<gIP, blk, 0, stream>>>(ip_hid, Wk_ip, nullptr, IPT, CAD, 2, 1e-5f,
                                                  nullptr, nullptr, nullptr, ipk, nullptr, 0);
        gemm_head_kernel<<<gIP, blk, 0, stream>>>(ip_hid, Wv_ip, nullptr, IPT, CAD, 3, 1e-5f,
                                                  nullptr, nullptr, nullptr, nullptr, ipvT, IPT);
    }

    // self attention (writes out), then IP attention (accumulates)
    attn_kernel<<<gAttn, blk, 0, stream>>>(qf, kf, vT, out, SEQ, SEQ, 0);
    attn_kernel<<<gAttn, blk, 0, stream>>>(qf, ipk, ipvT, out, SEQ, IPT, 1);
}
